// Transformer_5849745457396
// MI455X (gfx1250) — matrix operand order repaired and checked by reading, not compiled, not run on hardware
//
#include <hip/hip_runtime.h>
#include <hip/hip_bf16.h>
#include <math.h>

// Problem constants (from reference)
#define LL     6
#define BBATCH 2048
#define SS     64
#define DD     256
#define HH     8
#define DHH    64
#define VV     100
#define INNERD 512      // H*DH
#define FFD    1024     // 4*D
#define SCALE_INV 0.125f

#define WST32  40       // staging col stride (halfs) for 32-row K chunks (80B: 16B aligned, bank step 20)
#define WST64  72       // staging col stride (halfs) for 64-row K chunks (144B: 16B aligned, bank step 36)

typedef __attribute__((ext_vector_type(16))) _Float16 v16h;
typedef __attribute__((ext_vector_type(2)))  _Float16 v2h;
typedef __attribute__((ext_vector_type(8)))  float    v8f;
typedef _Float16 h16;

// ---------------------------------------------------------------------------
// WMMA wrapper: D = A(16x32 f16) * B(32x16 f16) + C(16x16 f32)
// ---------------------------------------------------------------------------
__device__ __forceinline__ v8f wmma16(v16h a, v16h b, v8f c) {
  return __builtin_amdgcn_wmma_f32_16x16x32_f16(
      /*neg_a=*/false, a, /*neg_b=*/false, b,
      /*c_mod=*/(short)0, c, /*reuse_a=*/false, /*reuse_b=*/false);
}

// A fragment (16x32 f16), row-major LDS, row stride lda (halfs).
// lane: row = mt*16 + lane%16, hs = lane/16.
// VGPR i : k = hs*16 + 2i (contiguous order, matches frag_b_cm)
// => one contiguous 32B run per lane -> 2x ds_load_b128.
__device__ __forceinline__ v16h frag_a(const h16* buf, int row, int hs, int k0, int lda) {
  v16h a;
  const h16* p = buf + row * lda + k0;
#pragma unroll
  for (int i = 0; i < 8; ++i) {
    int k = hs * 16 + 2 * i;
    a[2 * i]     = p[k];
    a[2 * i + 1] = p[k + 1];
  }
  return a;
}

// B fragment (32x16) from column-major f16 LDS: buf[col*stride + k].
// lane: col = nt*16 + lane%16, hs = lane/16; k = k0 + hs*16 + 2i.
// 16 contiguous halfs per lane -> 2x ds_load_b128 (stride keeps 16B alignment).
__device__ __forceinline__ v16h frag_b_cm(const h16* buf, int col, int hs, int k0, int stride) {
  v16h b;
  const h16* p = buf + col * stride + k0 + hs * 16;
#pragma unroll
  for (int i = 0; i < 16; ++i) b[i] = p[i];
  return b;
}

// Stage rows [k0, k0+KC) x NC columns of fp32 weights into LDS f16 column-major.
// Row-pairs per lane -> v_cvt_pk_f16_f32 + single ds_store_b32 per element-pair.
// Coalesced: 32 lanes read 32 consecutive columns of one row per iteration.
template <typename F>
__device__ __forceinline__ void stage_w(h16* dst, int stride, const float* W, int ldw,
                                        int k0, int KC, int NC, int wave, int lane,
                                        F colmap) {
  const int ppw = KC >> 4;  // row-pairs per wave (8 waves)
#pragma unroll
  for (int r = 0; r < ppw; ++r) {
    int kk = (wave * ppw + r) * 2;   // wave is SGPR -> scalar address component
    const float* s0 = W + (size_t)(k0 + kk) * ldw;
    const float* s1 = s0 + ldw;
    for (int c = lane; c < NC; c += 32) {
      int cm = colmap(c);
      v2h pk;
      pk[0] = (h16)s0[cm];
      pk[1] = (h16)s1[cm];
      *(v2h*)(dst + c * stride + kk) = pk;
    }
  }
}

__device__ __forceinline__ float gelu_exact(float g) {
  return 0.5f * g * (1.0f + erff(g * 0.70710678118654752f));
}

// LayerNorm: 4 threads per row (all 8 waves active), cross-lane reduce via shfl_xor.
__device__ __forceinline__ void layer_norm_rows(const float* src, h16* dst,
                                                const float* g, const float* bta, int tid) {
  const int r = tid >> 2, sub = tid & 3;
  const float* sr = src + r * DD;
  float s = 0.f;
  for (int j = sub; j < DD; j += 4) s += sr[j];
  s += __shfl_xor(s, 1);
  s += __shfl_xor(s, 2);
  float m = s * (1.0f / DD);
  float v = 0.f;
  for (int j = sub; j < DD; j += 4) { float d = sr[j] - m; v += d * d; }
  v += __shfl_xor(v, 1);
  v += __shfl_xor(v, 2);
  float inv = rsqrtf(v * (1.0f / DD) + 1e-5f);
  h16* w = dst + r * DD;
  for (int j = sub; j < DD; j += 4) w[j] = (h16)((sr[j] - m) * inv * g[j] + bta[j]);
}

// ---------------------------------------------------------------------------
// Fused 6-layer transformer: one workgroup (8 wave32) per batch element.
// Residual stream stays in LDS across all layers; weights staged through LDS.
// ---------------------------------------------------------------------------
__global__ __launch_bounds__(256) void fused_transformer_kernel(
    const int* __restrict__ x, const float* __restrict__ E,
    const float* __restrict__ ln1_g, const float* __restrict__ ln1_b,
    const float* __restrict__ Wqkv, const float* __restrict__ Wout,
    const float* __restrict__ bout,
    const float* __restrict__ ln2_g, const float* __restrict__ ln2_b,
    const float* __restrict__ W1, const float* __restrict__ b1,
    const float* __restrict__ W2, const float* __restrict__ b2,
    float* __restrict__ out) {
  __shared__ float sh_h[SS * DD];            // residual stream fp32 (64 KB)
  __shared__ h16   sh_n[SS * DD];            // LN output f16 (32 KB)
  __shared__ h16   sh_q[SS * DHH];           // q / attn-out / GEGLU tile (8 KB)
  __shared__ h16   sh_k[SS * DHH];           // k  [seq][feat]  (8 KB)
  __shared__ h16   sh_v[SS * DHH];           // vT [feat][seq]  (8 KB)
  __shared__ h16   sh_p[SS * SS];            // softmax probs f16 (8 KB)
  __shared__ __align__(16) h16 sh_w[192 * WST64];  // weight staging (27 KB), aliased below
  float* sh_sf = (float*)sh_w;               // f32 scores alias (16 KB of 27 KB)

  const int b    = blockIdx.x;
  const int tid  = threadIdx.x;
  // wave index is uniform across the wave: force into an SGPR so tile-group
  // selection and staging row math become scalar (s_cmp/s_cbranch, not v_cmpx).
  const int wave = __builtin_amdgcn_readfirstlane(tid >> 5);
  const int lane = tid & 31;
  const int ln16 = lane & 15;
  const int hs   = lane >> 4;
  const int mt   = wave & 3;            // M tile owned by this wave (SGPR)
  const int row  = mt * 16 + ln16;      // A row for this lane
  const v8f vzero = {};

  // ---- embedding gather: h[s][c] = E[s][x[b][s]][c] ----
  for (int idx = tid; idx < SS * DD; idx += 256) {
    int s = idx >> 8, c = idx & 255;
    int tok = x[b * SS + s];
    sh_h[idx] = E[((size_t)s * VV + tok) * DD + c];
  }
  __syncthreads();

  for (int l = 0; l < LL; ++l) {
    const float* Wqkv_l = Wqkv + (size_t)l * DD * (3 * INNERD);
    const float* Wout_l = Wout + (size_t)l * INNERD * DD;
    const float* W1_l   = W1   + (size_t)l * DD * (2 * FFD);
    const float* W2_l   = W2   + (size_t)l * FFD * DD;
    const float* b1_l   = b1   + (size_t)l * (2 * FFD);

    // ================= attention =================
    layer_norm_rows(sh_h, sh_n, ln1_g + l * DD, ln1_b + l * DD, tid);
    __syncthreads();

    for (int hd = 0; hd < HH; ++hd) {
      const int qc = hd * DHH;

      // --- q|k|v = n @ Wqkv[:, head cols]  (64x256 · 256x192, staged K=64) ---
      {
        const int ntb = (wave >> 2) * 6;  // 12 n-tiles total, 6 per wave (SGPR)
        v8f acc[6];
#pragma unroll
        for (int j = 0; j < 6; ++j) acc[j] = vzero;
        for (int k0 = 0; k0 < DD; k0 += 64) {
          stage_w(sh_w, WST64, Wqkv_l, 3 * INNERD, k0, 64, 192, wave, lane,
                  [&](int c) { return (c >> 6) * INNERD + qc + (c & 63); });
          __syncthreads();
#pragma unroll
          for (int ks = 0; ks < 64; ks += 32) {
            v16h a = frag_a(sh_n, row, hs, k0 + ks, DD);
#pragma unroll
            for (int j = 0; j < 6; ++j) {
              v16h bb = frag_b_cm(sh_w, (ntb + j) * 16 + ln16, hs, ks, WST64);
              acc[j] = wmma16(a, bb, acc[j]);
            }
          }
          __syncthreads();
        }
        // write-out: tile group is SGPR-uniform -> scalar branch per buffer
#pragma unroll
        for (int j = 0; j < 6; ++j) {
          const int nt     = ntb + j;           // 0..11 (SGPR)
          const int whichT = nt >> 2;           // 0=q, 1=k, 2=v (SGPR)
          const int lc     = (nt & 3) * 16 + ln16;
          if (whichT == 0) {
#pragma unroll
            for (int r = 0; r < 8; ++r)
              sh_q[(mt * 16 + r + 8 * hs) * DHH + lc] = (h16)acc[j][r];   // Q [seq][feat]
          } else if (whichT == 1) {
#pragma unroll
            for (int r = 0; r < 8; ++r)
              sh_k[(mt * 16 + r + 8 * hs) * DHH + lc] = (h16)acc[j][r];   // K [seq][feat]
          } else {
#pragma unroll
            for (int r = 0; r < 8; ++r)
              sh_v[lc * SS + (mt * 16 + r + 8 * hs)] = (h16)acc[j][r];    // V^T [feat][seq]
          }
        }
      }
      __syncthreads();

      // --- scores = (Q @ K^T)/8 : B[k=feat][n=key] = sh_k[key][feat] ---
      {
        const int ntb = (wave >> 2) * 2;
        v8f acc[2];
#pragma unroll
        for (int j = 0; j < 2; ++j) acc[j] = vzero;
#pragma unroll
        for (int k0 = 0; k0 < DHH; k0 += 32) {
          v16h a = frag_a(sh_q, row, hs, k0, DHH);
#pragma unroll
          for (int j = 0; j < 2; ++j) {
            v16h bb = frag_b_cm(sh_k, (ntb + j) * 16 + ln16, hs, k0, DHH);
            acc[j] = wmma16(a, bb, acc[j]);
          }
        }
#pragma unroll
        for (int j = 0; j < 2; ++j) {
          int col = (ntb + j) * 16 + ln16;
#pragma unroll
          for (int r = 0; r < 8; ++r)
            sh_sf[(mt * 16 + r + 8 * hs) * SS + col] = acc[j][r] * SCALE_INV;
        }
      }
      __syncthreads();

      // --- row softmax -> f16 probs (4 threads per row, all waves active) ---
      {
        const int r = tid >> 2, sub = tid & 3;
        float* srow = sh_sf + r * SS;
        float mx = -1e30f;
        for (int j = sub; j < SS; j += 4) mx = fmaxf(mx, srow[j]);
        mx = fmaxf(mx, __shfl_xor(mx, 1));
        mx = fmaxf(mx, __shfl_xor(mx, 2));
        float sum = 0.f;
        for (int j = sub; j < SS; j += 4) {
          float e = __expf(srow[j] - mx);
          srow[j] = e;
          sum += e;
        }
        sum += __shfl_xor(sum, 1);
        sum += __shfl_xor(sum, 2);
        float inv = 1.f / sum;
        for (int j = sub; j < SS; j += 4)
          sh_p[r * SS + j] = (h16)(srow[j] * inv);
      }
      __syncthreads();

      // --- o = P @ V : B[k=seq][n=feat] = sh_v[feat][seq] (col-major) ---
      {
        const int ntb = (wave >> 2) * 2;
        v8f acc[2];
#pragma unroll
        for (int j = 0; j < 2; ++j) acc[j] = vzero;
#pragma unroll
        for (int k0 = 0; k0 < SS; k0 += 32) {
          v16h a = frag_a(sh_p, row, hs, k0, SS);
#pragma unroll
          for (int j = 0; j < 2; ++j) {
            v16h bb = frag_b_cm(sh_v, (ntb + j) * 16 + ln16, hs, k0, SS);
            acc[j] = wmma16(a, bb, acc[j]);
          }
        }
#pragma unroll
        for (int j = 0; j < 2; ++j) {
          int col = (ntb + j) * 16 + ln16;
#pragma unroll
          for (int r = 0; r < 8; ++r)
            sh_q[(mt * 16 + r + 8 * hs) * DHH + col] = (h16)acc[j][r];  // o over q
        }
      }
      __syncthreads();

      // --- h += o @ Wout[head rows]  (64x64 · 64x256, staged K=32) ---
      {
        const int ntb = (wave >> 2) * 8;  // 16 n-tiles, 8 per wave
        const float* Wo = Wout_l + (size_t)hd * DHH * DD;
        v8f acc[8];
#pragma unroll
        for (int j = 0; j < 8; ++j) acc[j] = vzero;
        for (int k0 = 0; k0 < DHH; k0 += 32) {
          stage_w(sh_w, WST32, Wo, DD, k0, 32, 256, wave, lane,
                  [&](int c) { return c; });
          __syncthreads();
          v16h a = frag_a(sh_q, row, hs, k0, DHH);
#pragma unroll
          for (int j = 0; j < 8; ++j) {
            v16h bb = frag_b_cm(sh_w, (ntb + j) * 16 + ln16, hs, 0, WST32);
            acc[j] = wmma16(a, bb, acc[j]);
          }
          __syncthreads();
        }
#pragma unroll
        for (int j = 0; j < 8; ++j) {
          int col = (ntb + j) * 16 + ln16;
#pragma unroll
          for (int r = 0; r < 8; ++r)
            sh_h[(mt * 16 + r + 8 * hs) * DD + col] += acc[j][r];
        }
      }
      __syncthreads();
    } // heads

    // + bout
    for (int idx = tid; idx < SS * DD; idx += 256)
      sh_h[idx] += bout[l * DD + (idx & 255)];
    __syncthreads();

    // ================= FFN (GEGLU) =================
    layer_norm_rows(sh_h, sh_n, ln2_g + l * DD, ln2_b + l * DD, tid);
    __syncthreads();

    for (int jc = 0; jc < FFD; jc += DHH) {
      // --- a,g = n2 @ W1 chunk (64x256 · 256x128, staged K=64); ag -> sh_q ---
      {
        const int ntc = (wave >> 2) * 2;  // 4 col-chunks of 16 across waves
        v8f accA[2], accG[2];
#pragma unroll
        for (int j = 0; j < 2; ++j) { accA[j] = vzero; accG[j] = vzero; }
        for (int k0 = 0; k0 < DD; k0 += 64) {
          stage_w(sh_w, WST64, W1_l, 2 * FFD, k0, 64, 128, wave, lane,
                  [&](int c) { return (c < 64) ? (jc + c) : (FFD + jc + (c - 64)); });
          __syncthreads();
#pragma unroll
          for (int ks = 0; ks < 64; ks += 32) {
            v16h a = frag_a(sh_n, row, hs, k0 + ks, DD);
#pragma unroll
            for (int j = 0; j < 2; ++j) {
              int cA = (ntc + j) * 16 + ln16;  // 0..63
              v16h bA = frag_b_cm(sh_w, cA,      hs, ks, WST64);
              v16h bG = frag_b_cm(sh_w, 64 + cA, hs, ks, WST64);
              accA[j] = wmma16(a, bA, accA[j]);
              accG[j] = wmma16(a, bG, accG[j]);
            }
          }
          __syncthreads();
        }
#pragma unroll
        for (int j = 0; j < 2; ++j) {
          int cA = (ntc + j) * 16 + ln16;
          float ba = b1_l[jc + cA];
          float bg = b1_l[FFD + jc + cA];
#pragma unroll
          for (int r = 0; r < 8; ++r) {
            int m = mt * 16 + r + 8 * hs;
            float av = accA[j][r] + ba;
            float gv = accG[j][r] + bg;
            sh_q[m * DHH + cA] = (h16)(av * gelu_exact(gv));
          }
        }
      }
      __syncthreads();

      // --- h += ag @ W2[jc rows]  (64x64 · 64x256, staged K=32) ---
      {
        const int ntb = (wave >> 2) * 8;
        const float* W2c = W2_l + (size_t)jc * DD;
        v8f acc[8];
#pragma unroll
        for (int j = 0; j < 8; ++j) acc[j] = vzero;
        for (int k0 = 0; k0 < DHH; k0 += 32) {
          stage_w(sh_w, WST32, W2c, DD, k0, 32, 256, wave, lane,
                  [&](int c) { return c; });
          __syncthreads();
          v16h a = frag_a(sh_q, row, hs, k0, DHH);
#pragma unroll
          for (int j = 0; j < 8; ++j) {
            v16h bb = frag_b_cm(sh_w, (ntb + j) * 16 + ln16, hs, 0, WST32);
            acc[j] = wmma16(a, bb, acc[j]);
          }
          __syncthreads();
        }
#pragma unroll
        for (int j = 0; j < 8; ++j) {
          int col = (ntb + j) * 16 + ln16;
#pragma unroll
          for (int r = 0; r < 8; ++r)
            sh_h[(mt * 16 + r + 8 * hs) * DD + col] += acc[j][r];
        }
      }
      __syncthreads();
    }

    // + b2
    for (int idx = tid; idx < SS * DD; idx += 256)
      sh_h[idx] += b2[l * DD + (idx & 255)];
    __syncthreads();
  } // layers

  // write result
  float* outp = out + (size_t)b * SS * DD;
  for (int idx = tid; idx < SS * DD; idx += 256) outp[idx] = sh_h[idx];
}

// ---------------------------------------------------------------------------
extern "C" void kernel_launch(void* const* d_in, const int* in_sizes, int n_in,
                              void* d_out, int out_size, void* d_ws, size_t ws_size,
                              hipStream_t stream) {
  (void)in_sizes; (void)n_in; (void)out_size; (void)d_ws; (void)ws_size;
  const int*   x     = (const int*)  d_in[0];
  const float* E     = (const float*)d_in[1];
  const float* ln1_g = (const float*)d_in[2];
  const float* ln1_b = (const float*)d_in[3];
  const float* Wqkv  = (const float*)d_in[4];
  const float* Wout  = (const float*)d_in[5];
  const float* boutp = (const float*)d_in[6];
  const float* ln2_g = (const float*)d_in[7];
  const float* ln2_b = (const float*)d_in[8];
  const float* W1    = (const float*)d_in[9];
  const float* b1    = (const float*)d_in[10];
  const float* W2    = (const float*)d_in[11];
  const float* b2    = (const float*)d_in[12];
  float* out = (float*)d_out;

  fused_transformer_kernel<<<BBATCH, 256, 0, stream>>>(
      x, E, ln1_g, ln1_b, Wqkv, Wout, boutp, ln2_g, ln2_b, W1, b1, W2, b2, out);
}
